// SlotAttention_21990232556014
// MI455X (gfx1250) — matrix operand order repaired and checked by reading, not compiled, not run on hardware
//
#include <hip/hip_runtime.h>
#include <cstdint>

typedef __bf16 bf16_t;
typedef __attribute__((ext_vector_type(16))) __bf16 bf16x16;
typedef __attribute__((ext_vector_type(8)))  __bf16 bf16x8;
typedef __attribute__((ext_vector_type(8)))  float  f32x8;

#define LNP 520   // padded LDS row stride (bf16 elems), keeps 16B alignment

__device__ __forceinline__ f32x8 zero8() {
  f32x8 z;
#pragma unroll
  for (int i = 0; i < 8; ++i) z[i] = 0.f;
  return z;
}

__device__ __forceinline__ f32x8 wmma_bf16(bf16x16 a, bf16x16 b, f32x8 c) {
  return __builtin_amdgcn_wmma_f32_16x16x32_bf16(false, a, false, b, (short)0, c, false, false);
}

// Async DMA: 16B global -> LDS, tracked by ASYNCcnt (no VGPR round-trip).
__device__ __forceinline__ void async_ld_b128(void* lds, const void* g) {
  unsigned l = (unsigned)(uintptr_t)lds;                 // LDS aperture: low 32 bits = LDS offset
  unsigned long long ga = (unsigned long long)(uintptr_t)g;
  asm volatile("global_load_async_to_lds_b128 %0, %1, off"
               :: "v"(l), "v"(ga) : "memory");
}
__device__ __forceinline__ void wait_async0() {
  asm volatile("s_wait_asynccnt 0x0" ::: "memory");
}

// A fragment (16x32, bf16) from row-major [*, lda] at (mbase, kbase).
// lane L: m = mbase + L%16 ; elems 0..15 = K {0..15}+16*half
__device__ __forceinline__ bf16x16 load_A(const bf16_t* __restrict__ A, int lda,
                                          int mbase, int kbase, int lane) {
  const bf16_t* row = A + (size_t)(mbase + (lane & 15)) * lda + kbase + ((lane >> 4) << 4);
  bf16x8 lo = *(const bf16x8*)(row);
  bf16x8 hi = *(const bf16x8*)(row + 8);
  return __builtin_shufflevector(lo, hi, 0,1,2,3,4,5,6,7,8,9,10,11,12,13,14,15);
}

// B fragment (32x16) for y = x @ W^T : B[k][n] = W[nbase+n][kbase+k], W row-major [*, ldw].
// lane L: n = nbase + L%16 ; 16 contiguous K values starting at kbase + 16*half (one 32B load)
__device__ __forceinline__ bf16x16 load_B(const bf16_t* __restrict__ W, int ldw,
                                          int nbase, int kbase, int lane) {
  const bf16_t* row = W + (size_t)(nbase + (lane & 15)) * ldw + kbase + ((lane >> 4) << 4);
  return *(const bf16x16*)(row);
}

__device__ __forceinline__ void wave_red2(float& s, float& s2) {
#pragma unroll
  for (int o = 16; o > 0; o >>= 1) { s += __shfl_xor(s, o); s2 += __shfl_xor(s2, o); }
}

// ---------------- elementwise helpers ----------------
__global__ void __launch_bounds__(256) cvt_bf16_k(const float* __restrict__ in,
                                                  bf16_t* __restrict__ out, int n) {
  int i = blockIdx.x * 256 + threadIdx.x;
  if (i < n) out[i] = (bf16_t)in[i];
}

__global__ void __launch_bounds__(256) init_slots_k(const float* __restrict__ noise,
                                                    const float* __restrict__ mu,
                                                    const float* __restrict__ logsig,
                                                    float* __restrict__ slots) {
  int i = blockIdx.x * 256 + threadIdx.x;  // B*NS*512 = 524288 total
  int d = i & 511;
  slots[i] = mu[d] + __expf(logsig[d]) * noise[i];
}

// ---------------- LN(inputs) -> K (row-major bf16) and V^T (bf16) ----------------
// grid: B*N/32 = 4096 blocks, 256 threads (8 waves). Block handles 32 tokens.
__global__ void __launch_bounds__(256) ln_kv_k(const float* __restrict__ inputs,
                                               const float* __restrict__ g,
                                               const float* __restrict__ be,
                                               const bf16_t* __restrict__ Wk, const float* __restrict__ bk,
                                               const bf16_t* __restrict__ Wv, const float* __restrict__ bv,
                                               bf16_t* __restrict__ kout, bf16_t* __restrict__ vT) {
  __shared__ bf16_t xs[32 * LNP];
  const int tid = threadIdx.x, wave = tid >> 5, lane = tid & 31;
  const int base = blockIdx.x * 32;       // global token row
  const int b = base >> 10;               // batch (N=1024 divides base blocks)
  const int j0 = base & 1023;             // token offset within batch

  // LayerNorm: each wave normalizes 4 rows (8 waves * 4 = 32)
  for (int rr = 0; rr < 4; ++rr) {
    int r = wave * 4 + rr;
    const float* x = inputs + (size_t)(base + r) * 512;
    float vals[16], s = 0.f, s2 = 0.f;
#pragma unroll
    for (int i = 0; i < 16; ++i) { float t = x[lane + 32 * i]; vals[i] = t; s += t; s2 += t * t; }
    wave_red2(s, s2);
    float mean = s * (1.f / 512.f);
    float inv = rsqrtf(s2 * (1.f / 512.f) - mean * mean + 1e-5f);
#pragma unroll
    for (int i = 0; i < 16; ++i) {
      int c = lane + 32 * i;
      xs[r * LNP + c] = (bf16_t)(((vals[i] - mean) * inv) * g[c] + be[c]);
    }
  }
  __syncthreads();

  const int nb = wave * 64;  // each wave: 64 output columns, 2 row tiles
  // -------- K projection --------
  f32x8 acc[2][4];
#pragma unroll
  for (int mt = 0; mt < 2; ++mt)
#pragma unroll
    for (int nt = 0; nt < 4; ++nt) acc[mt][nt] = zero8();
  for (int kk = 0; kk < 512; kk += 32) {
    bf16x16 bfr[4];
#pragma unroll
    for (int nt = 0; nt < 4; ++nt) bfr[nt] = load_B(Wk, 512, nb + 16 * nt, kk, lane);
#pragma unroll
    for (int mt = 0; mt < 2; ++mt) {
      bf16x16 afr = load_A(xs, LNP, 16 * mt, kk, lane);
#pragma unroll
      for (int nt = 0; nt < 4; ++nt) acc[mt][nt] = wmma_bf16(afr, bfr[nt], acc[mt][nt]);
    }
  }
#pragma unroll
  for (int mt = 0; mt < 2; ++mt)
#pragma unroll
    for (int nt = 0; nt < 4; ++nt) {
      int n = nb + 16 * nt + (lane & 15);
      float bb = bk[n];
#pragma unroll
      for (int r = 0; r < 8; ++r) {
        int m = 16 * mt + r + ((lane >> 4) << 3);
        kout[(size_t)(base + m) * 512 + n] = (bf16_t)(acc[mt][nt][r] + bb);
      }
    }

  // -------- V projection (stored transposed: vT[b][d][j]) --------
#pragma unroll
  for (int mt = 0; mt < 2; ++mt)
#pragma unroll
    for (int nt = 0; nt < 4; ++nt) acc[mt][nt] = zero8();
  for (int kk = 0; kk < 512; kk += 32) {
    bf16x16 bfr[4];
#pragma unroll
    for (int nt = 0; nt < 4; ++nt) bfr[nt] = load_B(Wv, 512, nb + 16 * nt, kk, lane);
#pragma unroll
    for (int mt = 0; mt < 2; ++mt) {
      bf16x16 afr = load_A(xs, LNP, 16 * mt, kk, lane);
#pragma unroll
      for (int nt = 0; nt < 4; ++nt) acc[mt][nt] = wmma_bf16(afr, bfr[nt], acc[mt][nt]);
    }
  }
#pragma unroll
  for (int mt = 0; mt < 2; ++mt)
#pragma unroll
    for (int nt = 0; nt < 4; ++nt) {
      int n = nb + 16 * nt + (lane & 15);   // n = hidden dim d
      float bb = bv[n];
#pragma unroll
      for (int r = 0; r < 8; ++r) {
        int m = 16 * mt + r + ((lane >> 4) << 3);  // token within block
        vT[((size_t)b * 512 + n) * 1024 + (j0 + m)] = (bf16_t)(acc[mt][nt][r] + bb);
      }
    }
}

// ---------------- LN(slots) -> q (bf16) ; also snapshot prev slots as bf16 ----------------
// grid: 1024/32 = 32 blocks
__global__ void __launch_bounds__(256) ln_q_k(const float* __restrict__ slots,
                                              const float* __restrict__ g,
                                              const float* __restrict__ be,
                                              const bf16_t* __restrict__ Wq, const float* __restrict__ bq,
                                              bf16_t* __restrict__ qbf, bf16_t* __restrict__ slotsbf) {
  __shared__ bf16_t xs[32 * LNP];
  const int tid = threadIdx.x, wave = tid >> 5, lane = tid & 31;
  const int base = blockIdx.x * 32;

  for (int rr = 0; rr < 4; ++rr) {
    int r = wave * 4 + rr;
    const float* x = slots + (size_t)(base + r) * 512;
    float vals[16], s = 0.f, s2 = 0.f;
#pragma unroll
    for (int i = 0; i < 16; ++i) { float t = x[lane + 32 * i]; vals[i] = t; s += t; s2 += t * t; }
    wave_red2(s, s2);
    float mean = s * (1.f / 512.f);
    float inv = rsqrtf(s2 * (1.f / 512.f) - mean * mean + 1e-5f);
#pragma unroll
    for (int i = 0; i < 16; ++i) {
      int c = lane + 32 * i;
      slotsbf[(size_t)(base + r) * 512 + c] = (bf16_t)vals[i];          // raw prev slots
      xs[r * LNP + c] = (bf16_t)(((vals[i] - mean) * inv) * g[c] + be[c]);
    }
  }
  __syncthreads();

  const int nb = wave * 64;
  f32x8 acc[2][4];
#pragma unroll
  for (int mt = 0; mt < 2; ++mt)
#pragma unroll
    for (int nt = 0; nt < 4; ++nt) acc[mt][nt] = zero8();
  for (int kk = 0; kk < 512; kk += 32) {
    bf16x16 bfr[4];
#pragma unroll
    for (int nt = 0; nt < 4; ++nt) bfr[nt] = load_B(Wq, 512, nb + 16 * nt, kk, lane);
#pragma unroll
    for (int mt = 0; mt < 2; ++mt) {
      bf16x16 afr = load_A(xs, LNP, 16 * mt, kk, lane);
#pragma unroll
      for (int nt = 0; nt < 4; ++nt) acc[mt][nt] = wmma_bf16(afr, bfr[nt], acc[mt][nt]);
    }
  }
#pragma unroll
  for (int mt = 0; mt < 2; ++mt)
#pragma unroll
    for (int nt = 0; nt < 4; ++nt) {
      int n = nb + 16 * nt + (lane & 15);
      float bb = bq[n];
#pragma unroll
      for (int r = 0; r < 8; ++r) {
        int m = 16 * mt + r + ((lane >> 4) << 3);
        qbf[(size_t)(base + m) * 512 + n] = (bf16_t)(acc[mt][nt][r] + bb);
      }
    }
}

// ---------------- dots + slot-softmax + EPS + per-slot rowsum ----------------
// grid: B = 128 blocks (one per batch), 256 threads; wave handles 128 columns.
__global__ void __launch_bounds__(256) attn_k(const bf16_t* __restrict__ qbf,
                                              const bf16_t* __restrict__ kbf,
                                              bf16_t* __restrict__ attn,
                                              float* __restrict__ rowsum) {
  __shared__ bf16_t qs[16 * LNP];
  __shared__ float rs[8];
  const int tid = threadIdx.x, wave = tid >> 5, lane = tid & 31;
  const int b = blockIdx.x;

  // Stage q rows 0..7 via async global->LDS DMA (8 rows * 64 chunks of 16B).
  for (int c = tid; c < 512; c += 256) {
    int r = c >> 6;                 // 0..7
    int col = (c & 63) * 8;         // bf16 element offset (16B chunks)
    async_ld_b128(&qs[r * LNP + col], qbf + (size_t)(b * 8 + r) * 512 + col);
  }
  // Zero pad rows 8..15 so the WMMA A-matrix upper half is clean.
  for (int i = tid; i < 8 * 512; i += 256) {
    int r = 8 + (i >> 9), col = i & 511;
    qs[r * LNP + col] = (bf16_t)0.f;
  }
  if (tid < 8) rs[tid] = 0.f;
  wait_async0();
  __syncthreads();

  const int jb = wave * 128;
  const bf16_t* kb = kbf + (size_t)b * 1024 * 512;
  f32x8 acc[8];
#pragma unroll
  for (int nt = 0; nt < 8; ++nt) acc[nt] = zero8();
  for (int kk = 0; kk < 512; kk += 32) {
    bf16x16 afr = load_A(qs, LNP, 0, kk, lane);
#pragma unroll
    for (int nt = 0; nt < 8; ++nt)
      acc[nt] = wmma_bf16(afr, load_B(kb, 512, jb + 16 * nt, kk, lane), acc[nt]);
  }

  const float scale = 0.044194173824159216f;  // 512^-0.5
  const int half = lane >> 4, col = lane & 15;
  float ps[8];
#pragma unroll
  for (int r = 0; r < 8; ++r) ps[r] = 0.f;

#pragma unroll
  for (int nt = 0; nt < 8; ++nt) {
    int j = jb + 16 * nt + col;
    if (half == 0) {
      // lane holds all 8 slot logits of column j in its 8 accumulator regs
      float d[8], mx = -1e30f;
#pragma unroll
      for (int r = 0; r < 8; ++r) { d[r] = acc[nt][r] * scale; mx = fmaxf(mx, d[r]); }
      float se = 0.f;
#pragma unroll
      for (int r = 0; r < 8; ++r) { d[r] = __expf(d[r] - mx); se += d[r]; }
      float inv = 1.f / se;
#pragma unroll
      for (int r = 0; r < 8; ++r) {
        float a = d[r] * inv + 1e-8f;
        ps[r] += a;
        attn[((size_t)b * 16 + r) * 1024 + j] = (bf16_t)a;
      }
    } else {
      // pad rows 8..15 with zeros so the updates GEMM A-matrix is clean
#pragma unroll
      for (int r = 0; r < 8; ++r) attn[((size_t)b * 16 + 8 + r) * 1024 + j] = (bf16_t)0.f;
    }
  }
  // reduce per-slot partial sums across lanes 0..15 (xor masks stay inside the half)
#pragma unroll
  for (int o = 8; o > 0; o >>= 1) {
#pragma unroll
    for (int r = 0; r < 8; ++r) ps[r] += __shfl_xor(ps[r], o);
  }
  if (lane == 0) {
#pragma unroll
    for (int r = 0; r < 8; ++r) atomicAdd(&rs[r], ps[r]);
  }
  __syncthreads();
  if (tid < 8) rowsum[b * 8 + tid] = rs[tid];
}

// ---------------- updates = (attn / rowsum) @ v ----------------
// grid: B = 128 blocks; wave handles 64 hidden dims; K = N = 1024.
__global__ void __launch_bounds__(256) updates_k(const bf16_t* __restrict__ attn,
                                                 const bf16_t* __restrict__ vT,
                                                 const float* __restrict__ rowsum,
                                                 float* __restrict__ upd,
                                                 bf16_t* __restrict__ updbf) {
  const int tid = threadIdx.x, wave = tid >> 5, lane = tid & 31;
  const int b = blockIdx.x;
  const bf16_t* ab = attn + (size_t)b * 16 * 1024;
  const bf16_t* vb = vT + (size_t)b * 512 * 1024;
  const int nb = wave * 64;
  f32x8 acc[4];
#pragma unroll
  for (int nt = 0; nt < 4; ++nt) acc[nt] = zero8();
  for (int kk = 0; kk < 1024; kk += 32) {
    bf16x16 afr = load_A(ab, 1024, 0, kk, lane);
#pragma unroll
    for (int nt = 0; nt < 4; ++nt)
      acc[nt] = wmma_bf16(afr, load_B(vb, 1024, nb + 16 * nt, kk, lane), acc[nt]);
  }
  if ((lane >> 4) == 0) {   // only slot rows 0..7 are real
#pragma unroll
    for (int r = 0; r < 8; ++r) {
      float inv = 1.f / rowsum[b * 8 + r];
#pragma unroll
      for (int nt = 0; nt < 4; ++nt) {
        int n = nb + 16 * nt + (lane & 15);
        size_t idx = (size_t)(b * 8 + r) * 512 + n;
        float v = acc[nt][r] * inv;
        upd[idx] = v;
        updbf[idx] = (bf16_t)v;
      }
    }
  }
}

// ---- generic WMMA GEMM with async double-buffered A staging through LDS ----
// out = [relu](A @ W^T + bias) (+residual). grid: (N/128, M/64), 256 threads.
// A block (64x32 bf16) is DMA'd global->LDS (ASYNCcnt) and shared by all 8 waves,
// removing the 8x redundant global A reads; next block's DMA overlaps WMMAs.
#define GAP 40  // padded LDS row stride for the A block
__global__ void __launch_bounds__(256) gemm_k(const bf16_t* __restrict__ A,
                                              const bf16_t* __restrict__ W,
                                              const float* __restrict__ bias,
                                              const float* __restrict__ residual,
                                              float* __restrict__ outF,
                                              bf16_t* __restrict__ outB,
                                              int M, int N, int K, int relu) {
  __shared__ bf16_t As[2][64 * GAP];
  const int tid = threadIdx.x, wave = tid >> 5, lane = tid & 31;
  const int nb = blockIdx.x * 128 + wave * 16;
  const int mb = blockIdx.y * 64;
  const int srow = tid >> 2;          // 0..63: staging row
  const int sch = (tid & 3) * 8;      // 16B chunk within the 32-wide K block

  // prologue: stage first A block
  async_ld_b128(&As[0][srow * GAP + sch], A + (size_t)(mb + srow) * K + sch);

  f32x8 acc[4];
#pragma unroll
  for (int mt = 0; mt < 4; ++mt) acc[mt] = zero8();

  for (int kk = 0; kk < K; kk += 32) {
    int cur = (kk >> 5) & 1;
    wait_async0();          // own-wave DMA done
    __syncthreads();        // all waves' DMA done; all waves done reading buf 'cur^1'
    if (kk + 32 < K)        // overlap next block's DMA with this block's WMMAs
      async_ld_b128(&As[cur ^ 1][srow * GAP + sch],
                    A + (size_t)(mb + srow) * K + kk + 32 + sch);
    __builtin_prefetch(W + (size_t)(nb + (lane & 15)) * K + kk + 32);
    bf16x16 bfr = load_B(W, K, nb, kk, lane);
#pragma unroll
    for (int mt = 0; mt < 4; ++mt) {
      bf16x16 afr = load_A(As[cur], GAP, 16 * mt, 0, lane);
      acc[mt] = wmma_bf16(afr, bfr, acc[mt]);
    }
  }
  const int n = nb + (lane & 15);
  const float bb = bias ? bias[n] : 0.f;
#pragma unroll
  for (int mt = 0; mt < 4; ++mt) {
#pragma unroll
    for (int r = 0; r < 8; ++r) {
      int m = mb + 16 * mt + r + ((lane >> 4) << 3);
      float v = acc[mt][r] + bb;
      if (relu) v = fmaxf(v, 0.f);
      size_t idx = (size_t)m * N + n;
      if (residual) v += residual[idx];
      if (outF) outF[idx] = v;
      if (outB) outB[idx] = (bf16_t)v;
    }
  }
}

// ---------------- fused GRU gate combine + LayerNorm -> p (bf16) ----------------
// grid: 1024/8 = 128 blocks; each wave owns one row of 512.
__global__ void __launch_bounds__(256) gru_ln_k(const float* __restrict__ xg,
                                                const float* __restrict__ hg,
                                                float* __restrict__ slots,
                                                const float* __restrict__ g,
                                                const float* __restrict__ be,
                                                bf16_t* __restrict__ pbf) {
  const int tid = threadIdx.x, wave = tid >> 5, lane = tid & 31;
  const int row = blockIdx.x * 8 + wave;
  const float* xr_ = xg + (size_t)row * 1536;
  const float* hr_ = hg + (size_t)row * 1536;
  float* sl = slots + (size_t)row * 512;
  float vals[16], s = 0.f, s2 = 0.f;
#pragma unroll
  for (int i = 0; i < 16; ++i) {
    int c = lane + 32 * i;
    float xr = xr_[c], xz = xr_[512 + c], xn = xr_[1024 + c];
    float hr = hr_[c], hz = hr_[512 + c], hn = hr_[1024 + c];
    float prev = sl[c];
    float rg = 1.f / (1.f + __expf(-(xr + hr)));
    float zg = 1.f / (1.f + __expf(-(xz + hz)));
    float ng = tanhf(xn + rg * hn);
    float sn = (1.f - zg) * ng + zg * prev;
    sl[c] = sn;
    vals[i] = sn; s += sn; s2 += sn * sn;
  }
  wave_red2(s, s2);
  float mean = s * (1.f / 512.f);
  float inv = rsqrtf(s2 * (1.f / 512.f) - mean * mean + 1e-5f);
#pragma unroll
  for (int i = 0; i < 16; ++i) {
    int c = lane + 32 * i;
    pbf[(size_t)row * 512 + c] = (bf16_t)(((vals[i] - mean) * inv) * g[c] + be[c]);
  }
}

// =====================================================================
extern "C" void kernel_launch(void* const* d_in, const int* in_sizes, int n_in,
                              void* d_out, int out_size, void* d_ws, size_t ws_size,
                              hipStream_t stream) {
  const float* inputs   = (const float*)d_in[0];
  const float* noise    = (const float*)d_in[1];
  const float* mu       = (const float*)d_in[2];
  const float* logsig   = (const float*)d_in[3];
  const float* g_in     = (const float*)d_in[4];
  const float* be_in    = (const float*)d_in[5];
  const float* g_sl     = (const float*)d_in[6];
  const float* be_sl    = (const float*)d_in[7];
  const float* g_ff     = (const float*)d_in[8];
  const float* be_ff    = (const float*)d_in[9];
  const float* Wq       = (const float*)d_in[10];
  const float* bq       = (const float*)d_in[11];
  const float* Wk       = (const float*)d_in[12];
  const float* bk       = (const float*)d_in[13];
  const float* Wv       = (const float*)d_in[14];
  const float* bv       = (const float*)d_in[15];
  const float* W_ih     = (const float*)d_in[16];
  const float* b_ih     = (const float*)d_in[17];
  const float* W_hh     = (const float*)d_in[18];
  const float* b_hh     = (const float*)d_in[19];
  const float* W1       = (const float*)d_in[20];
  const float* b1       = (const float*)d_in[21];
  const float* W2       = (const float*)d_in[22];
  const float* b2       = (const float*)d_in[23];
  float* out = (float*)d_out;

  // ----- workspace carving -----
  char* ws = (char*)d_ws;
  size_t off = 0;
  auto alloc = [&](size_t bytes) -> char* {
    char* p = ws + off;
    off += (bytes + 255) & ~(size_t)255;
    return p;
  };
  const size_t BN = 128 * 1024;           // tokens
  bf16_t* kbf    = (bf16_t*)alloc(BN * 512 * 2);        // [B*N, 512]
  bf16_t* vT     = (bf16_t*)alloc(BN * 512 * 2);        // [B, 512, N]
  float*  slots  = (float*)alloc(1024 * 512 * 4);       // [B*NS, 512]
  bf16_t* slotsbf= (bf16_t*)alloc(1024 * 512 * 2);
  bf16_t* qbf    = (bf16_t*)alloc(1024 * 512 * 2);
  bf16_t* attn   = (bf16_t*)alloc((size_t)128 * 16 * 1024 * 2);
  float*  rowsum = (float*)alloc(1024 * 4);
  float*  upd    = (float*)alloc(1024 * 512 * 4);
  bf16_t* updbf  = (bf16_t*)alloc(1024 * 512 * 2);
  float*  xg     = (float*)alloc((size_t)1024 * 1536 * 4);
  float*  hg     = (float*)alloc((size_t)1024 * 1536 * 4);
  bf16_t* pbf    = (bf16_t*)alloc(1024 * 512 * 2);
  bf16_t* h1bf   = (bf16_t*)alloc(1024 * 512 * 2);
  bf16_t* Wq_b   = (bf16_t*)alloc(262144 * 2);
  bf16_t* Wk_b   = (bf16_t*)alloc(262144 * 2);
  bf16_t* Wv_b   = (bf16_t*)alloc(262144 * 2);
  bf16_t* Wih_b  = (bf16_t*)alloc(786432 * 2);
  bf16_t* Whh_b  = (bf16_t*)alloc(786432 * 2);
  bf16_t* W1_b   = (bf16_t*)alloc(262144 * 2);
  bf16_t* W2_b   = (bf16_t*)alloc(262144 * 2);

  // ----- weight conversion (fp32 -> bf16) -----
  cvt_bf16_k<<<262144 / 256, 256, 0, stream>>>(Wq, Wq_b, 262144);
  cvt_bf16_k<<<262144 / 256, 256, 0, stream>>>(Wk, Wk_b, 262144);
  cvt_bf16_k<<<262144 / 256, 256, 0, stream>>>(Wv, Wv_b, 262144);
  cvt_bf16_k<<<786432 / 256, 256, 0, stream>>>(W_ih, Wih_b, 786432);
  cvt_bf16_k<<<786432 / 256, 256, 0, stream>>>(W_hh, Whh_b, 786432);
  cvt_bf16_k<<<262144 / 256, 256, 0, stream>>>(W1, W1_b, 262144);
  cvt_bf16_k<<<262144 / 256, 256, 0, stream>>>(W2, W2_b, 262144);

  // ----- slot init -----
  init_slots_k<<<524288 / 256, 256, 0, stream>>>(noise, mu, logsig, slots);

  // ----- LN(inputs) + K/V projections -----
  ln_kv_k<<<(int)(BN / 32), 256, 0, stream>>>(inputs, g_in, be_in,
                                              Wk_b, bk, Wv_b, bv, kbf, vT);

  // ----- 3 slot-attention iterations -----
  for (int it = 0; it < 3; ++it) {
    ln_q_k<<<32, 256, 0, stream>>>(slots, g_sl, be_sl, Wq_b, bq, qbf, slotsbf);
    attn_k<<<128, 256, 0, stream>>>(qbf, kbf, attn, rowsum);
    updates_k<<<128, 256, 0, stream>>>(attn, vT, rowsum, upd, updbf);
    // GRU gate GEMMs: [1024,1536] = [1024,512] @ [1536,512]^T
    gemm_k<<<dim3(1536 / 128, 1024 / 64), 256, 0, stream>>>(
        updbf, Wih_b, b_ih, nullptr, xg, nullptr, 1024, 1536, 512, 0);
    gemm_k<<<dim3(1536 / 128, 1024 / 64), 256, 0, stream>>>(
        slotsbf, Whh_b, b_hh, nullptr, hg, nullptr, 1024, 1536, 512, 0);
    gru_ln_k<<<128, 256, 0, stream>>>(xg, hg, slots, g_ff, be_ff, pbf);
    // MLP: h1 = relu(p @ W1^T + b1)  -> bf16
    gemm_k<<<dim3(512 / 128, 1024 / 64), 256, 0, stream>>>(
        pbf, W1_b, b1, nullptr, nullptr, h1bf, 1024, 512, 512, 1);
    // slots = slots + h1 @ W2^T + b2 (final iteration writes d_out)
    float* outp = (it == 2) ? out : slots;
    gemm_k<<<dim3(512 / 128, 1024 / 64), 256, 0, stream>>>(
        h1bf, W2_b, b2, slots, outp, nullptr, 1024, 512, 512, 0);
  }
}
